// GCN_64836826300609
// MI455X (gfx1250) — matrix operand order repaired and checked by reading, not compiled, not run on hardware
//
#include <hip/hip_runtime.h>
#include <hip/hip_bf16.h>
#include <cstdint>
#include <cstddef>

#define N_NODES  100000
#define N_GRAPHS 512
#define HID      128
#define N_EDGES  1600000

typedef __attribute__((ext_vector_type(16))) __bf16          v16bf;
typedef __attribute__((ext_vector_type(16))) unsigned short  v16u;
typedef __attribute__((ext_vector_type(8)))  float           v8f;

__device__ __forceinline__ unsigned short f2bf(float f) {
    unsigned int u = __float_as_uint(f);
    u += 0x7FFFu + ((u >> 16) & 1u);          // round-to-nearest-even
    return (unsigned short)(u >> 16);
}

// low 32 bits of a generic LDS pointer == LDS byte offset (aperture mapping)
__device__ __forceinline__ unsigned lds_off(const void* p) {
    return (unsigned)(size_t)p;
}

// ---------------------------------------------------------------------------
// One-time conversion of all weight matrices to bf16.
// Layout in out[]: W1 @0, W2 @16384, W3 @32768, Wc1 @49152 (256x128)
// ---------------------------------------------------------------------------
__global__ void convert_weights_kernel(const float* __restrict__ W1, const float* __restrict__ W2,
                                       const float* __restrict__ W3, const float* __restrict__ Wc1,
                                       unsigned short* __restrict__ out) {
    const int n1 = 128 * 128;
    const int total = 3 * n1 + 256 * 128;
    int i = blockIdx.x * blockDim.x + threadIdx.x;
    if (i >= total) return;
    float v;
    if      (i <     n1) v = W1[i];
    else if (i < 2 * n1) v = W2[i - n1];
    else if (i < 3 * n1) v = W3[i - 2 * n1];
    else                 v = Wc1[i - 3 * n1];
    out[i] = f2bf(v);
}

// ---------------------------------------------------------------------------
// WMMA GEMM: out[M,128] = A[M,K] @ W[K,128]  (+bias, optional relu)
// 256 threads = 8 waves; block computes 128 rows x 128 cols.
// Wave w handles M-tile w (16 rows) across all 8 N-tiles.
// W (already bf16 in global) is staged to LDS with global_load_async_to_lds_b128;
// A tile is converted f32->bf16 through registers into LDS.
// Dynamic LDS: W bf16 [K][128] then A bf16 [128][K].
// ---------------------------------------------------------------------------
__global__ void gemm128_wmma(const float* __restrict__ A, const unsigned short* __restrict__ Wbf,
                             const float* __restrict__ bias, float* __restrict__ out,
                             int M, int K, int relu) {
    extern __shared__ unsigned short smem[];
    unsigned short* wl = smem;            // [K][128] bf16
    unsigned short* al = smem + K * 128;  // [128][K] bf16

    const int tid = threadIdx.x;
    const int r0  = blockIdx.x * 128;

    // --- async copy of bf16 weights into LDS (per-lane 16B chunks) ---
    {
        const char* wsrc = (const char*)Wbf;
        const unsigned lbase = lds_off(wl);
        const int wbytes = K * 128 * 2;
        for (int o = tid * 16; o < wbytes; o += 256 * 16) {
            unsigned la = lbase + (unsigned)o;
            unsigned long long ga = (unsigned long long)(wsrc + o);
            asm volatile("global_load_async_to_lds_b128 %0, %1, off"
                         :: "v"(la), "v"(ga) : "memory");
        }
    }

    // --- stage A tile (convert f32 -> bf16) ---
    for (int i = tid; i < 128 * K; i += 256) {
        int row = i / K, k = i - row * K;
        int gr  = r0 + row;
        al[i] = (gr < M) ? f2bf(A[(size_t)gr * K + k]) : (unsigned short)0;
    }

    asm volatile("s_wait_asynccnt 0" ::: "memory");
    __syncthreads();

    const int wave = tid >> 5, lane = tid & 31;
    const int mt   = wave;
    const int half = lane >> 4, l16 = lane & 15;

    v8f acc[8];
    #pragma unroll
    for (int nt = 0; nt < 8; nt++)
        #pragma unroll
        for (int t = 0; t < 8; t++) acc[nt][t] = 0.0f;

    const unsigned short* arow = al + (size_t)(mt * 16 + l16) * K;
    const int ksteps = K >> 5;

    for (int ks = 0; ks < ksteps; ks++) {
        // A fragment (16x32 bf16): elems 0..7 -> K = half*8 + j, elems 8..15 -> K = 16 + half*8 + j
        v16u af;
        const int kb = ks * 32 + half * 8;
        #pragma unroll
        for (int j = 0; j < 8; j++) {
            af[j]     = arow[kb + j];
            af[8 + j] = arow[kb + 16 + j];
        }
        v16bf av = __builtin_bit_cast(v16bf, af);

        const unsigned short* wrow = wl + (size_t)(ks * 32 + half * 8) * 128 + l16;
        #pragma unroll
        for (int nt = 0; nt < 8; nt++) {
            // B fragment (32x16 bf16): N = nt*16 + lane%16, elems 0..7 -> K = half*8 + j, elems 8..15 -> K = 16 + half*8 + j
            v16u bfr;
            #pragma unroll
            for (int j = 0; j < 8; j++) { bfr[j] = wrow[j * 128 + nt * 16]; bfr[8 + j] = wrow[(16 + j) * 128 + nt * 16]; }
            v16bf bv = __builtin_bit_cast(v16bf, bfr);
            acc[nt] = __builtin_amdgcn_wmma_f32_16x16x32_bf16(
                false, av, false, bv, (short)0, acc[nt], false, false);
        }
    }

    // D layout: lane l, VGPR r -> M = r + 8*(l/16), N = l%16
    #pragma unroll
    for (int nt = 0; nt < 8; nt++) {
        const int gcol = nt * 16 + l16;
        #pragma unroll
        for (int r = 0; r < 8; r++) {
            const int grow = r0 + mt * 16 + r + half * 8;
            if (grow < M) {
                float v = acc[nt][r];
                if (bias) v += bias[gcol];
                if (relu) v = fmaxf(v, 0.0f);
                out[(size_t)grow * 128 + gcol] = v;
            }
        }
    }
}

// ---------------------------------------------------------------------------
// Graph kernels
// ---------------------------------------------------------------------------
__global__ void deg_kernel(const int* __restrict__ dst, float* __restrict__ deg, int ne) {
    int i = blockIdx.x * blockDim.x + threadIdx.x;
    if (i < ne) atomicAdd(&deg[dst[i]], 1.0f);
}

__global__ void dinv_kernel(const float* __restrict__ deg, float* __restrict__ dinv, int n) {
    int i = blockIdx.x * blockDim.x + threadIdx.x;
    if (i < n) dinv[i] = rsqrtf(deg[i] + 1.0f);   // +1 self-loop
}

// one wave per edge: lane gathers float4 of h[src], atomically adds to agg[dst]
__global__ void edge_agg_kernel(const int* __restrict__ src, const int* __restrict__ dst,
                                const float* __restrict__ dinv, const float* __restrict__ h,
                                float* __restrict__ agg, int ne) {
    int gt   = blockIdx.x * blockDim.x + threadIdx.x;
    int e    = gt >> 5;
    int lane = gt & 31;
    if (e >= ne) return;
    int s = src[e], d = dst[e];
    float norm = dinv[s] * dinv[d];
    float4 v = ((const float4*)(h + (size_t)s * 128))[lane];
    float* ap = agg + (size_t)d * 128 + lane * 4;
    atomicAdd(ap + 0, v.x * norm);
    atomicAdd(ap + 1, v.y * norm);
    atomicAdd(ap + 2, v.z * norm);
    atomicAdd(ap + 3, v.w * norm);
}

__global__ void selfloop_bias_kernel(const float* __restrict__ h, const float* __restrict__ dinv,
                                     const float* __restrict__ b, float* __restrict__ out,
                                     int n, int relu) {
    int i = blockIdx.x * blockDim.x + threadIdx.x;
    if (i >= n * 128) return;
    int node = i >> 7, c = i & 127;
    float di = dinv[node];
    float v = out[i] + h[i] * di * di + b[c];
    if (relu) v = fmaxf(v, 0.0f);
    out[i] = v;
}

__global__ void pool_init_kernel(float* __restrict__ sums, float* __restrict__ cnt,
                                 float* __restrict__ mx) {
    int i = blockIdx.x * blockDim.x + threadIdx.x;
    if (i < N_GRAPHS * HID) { sums[i] = 0.0f; mx[i] = -__builtin_inff(); }
    if (i < N_GRAPHS) cnt[i] = 0.0f;
}

__device__ __forceinline__ void atomicMaxF(float* addr, float val) {
    if (val >= 0.0f) atomicMax((int*)addr, __float_as_int(val));
    else             atomicMin((unsigned int*)addr, __float_as_uint(val));
}

__global__ void pool_kernel(const float* __restrict__ h, const int* __restrict__ batch,
                            float* __restrict__ sums, float* __restrict__ cnt,
                            float* __restrict__ mx, int n) {
    int node = blockIdx.x;
    int c    = threadIdx.x;
    if (node >= n) return;
    int b = batch[node];
    float v = h[(size_t)node * 128 + c];
    atomicAdd(&sums[b * 128 + c], v);
    atomicMaxF(&mx[b * 128 + c], v);
    if (c == 0) atomicAdd(&cnt[b], 1.0f);
}

__global__ void pool_finalize_kernel(const float* __restrict__ sums, const float* __restrict__ cnt,
                                     const float* __restrict__ mx, float* __restrict__ g) {
    int i = blockIdx.x * blockDim.x + threadIdx.x;
    if (i >= N_GRAPHS * 256) return;
    int gi = i >> 8, c = i & 255;
    float v;
    if (c < 128) {
        v = sums[gi * 128 + c] / fmaxf(cnt[gi], 1.0f);
    } else {
        float m = mx[gi * 128 + (c - 128)];
        v = (m > -__builtin_inff()) ? m : 0.0f;   // empty graph -> 0
    }
    g[i] = v;
}

__global__ void final_linear_kernel(const float* __restrict__ c1, const float* __restrict__ W,
                                    const float* __restrict__ b, float* __restrict__ out) {
    int i = blockIdx.x * blockDim.x + threadIdx.x;   // 512 * 10
    if (i >= N_GRAPHS * 10) return;
    int gi = i / 10, o = i - gi * 10;
    float acc = b[o];
    const float* row = c1 + (size_t)gi * 128;
    #pragma unroll 4
    for (int k = 0; k < 128; k++) acc += row[k] * W[k * 10 + o];
    out[i] = acc;
}

// ---------------------------------------------------------------------------
// Launch
// ---------------------------------------------------------------------------
extern "C" void kernel_launch(void* const* d_in, const int* in_sizes, int n_in,
                              void* d_out, int out_size, void* d_ws, size_t ws_size,
                              hipStream_t stream) {
    (void)in_sizes; (void)n_in; (void)out_size; (void)ws_size;

    const float* x    = (const float*)d_in[0];
    const int*   ei   = (const int*)d_in[1];
    const int*   batc = (const int*)d_in[2];
    const float* W1   = (const float*)d_in[3];  const float* b1  = (const float*)d_in[4];
    const float* W2   = (const float*)d_in[5];  const float* b2  = (const float*)d_in[6];
    const float* W3   = (const float*)d_in[7];  const float* b3  = (const float*)d_in[8];
    const float* Wc1  = (const float*)d_in[9];  const float* bc1 = (const float*)d_in[10];
    const float* Wc2  = (const float*)d_in[11]; const float* bc2 = (const float*)d_in[12];
    const int* src = ei;
    const int* dst = ei + N_EDGES;

    char* ws = (char*)d_ws;
    size_t off = 0;
    auto alloc = [&](size_t bytes) -> char* {
        char* p = ws + off;
        off = (off + bytes + 255) & ~(size_t)255;
        return p;
    };
    float*          buf1 = (float*)alloc((size_t)N_NODES * HID * 4);   // h = in @ W
    float*          buf2 = (float*)alloc((size_t)N_NODES * HID * 4);   // agg / layer output
    float*          deg  = (float*)alloc((size_t)N_NODES * 4);
    float*          dinv = (float*)alloc((size_t)N_NODES * 4);
    float*          sums = (float*)alloc((size_t)N_GRAPHS * HID * 4);
    float*          cnt  = (float*)alloc((size_t)N_GRAPHS * 4);
    float*          mx   = (float*)alloc((size_t)N_GRAPHS * HID * 4);
    float*          g    = (float*)alloc((size_t)N_GRAPHS * 256 * 4);
    float*          c1   = (float*)alloc((size_t)N_GRAPHS * HID * 4);
    unsigned short* wbf  = (unsigned short*)alloc((size_t)(3 * 128 * 128 + 256 * 128) * 2);

    const unsigned short* wbf1  = wbf;
    const unsigned short* wbf2  = wbf + 128 * 128;
    const unsigned short* wbf3  = wbf + 2 * 128 * 128;
    const unsigned short* wbfc1 = wbf + 3 * 128 * 128;

    // one-time prep: bf16 weights, degrees, normalization
    convert_weights_kernel<<<(3 * 128 * 128 + 256 * 128 + 255) / 256, 256, 0, stream>>>(
        W1, W2, W3, Wc1, wbf);
    hipMemsetAsync(deg, 0, (size_t)N_NODES * 4, stream);
    deg_kernel<<<(N_EDGES + 255) / 256, 256, 0, stream>>>(dst, deg, N_EDGES);
    dinv_kernel<<<(N_NODES + 255) / 256, 256, 0, stream>>>(deg, dinv, N_NODES);

    const int    gemm_blocks = (N_NODES + 127) / 128;
    const size_t lds128      = (size_t)(128 * 128 + 128 * 128) * 2;  // 64 KB
    const int    agg_blocks  = (int)(((size_t)N_EDGES * 32 + 255) / 256);
    const int    elem_blocks = (N_NODES * 128 + 255) / 256;

    const float*          layer_in[3] = { x, buf2, buf2 };
    const unsigned short* Ws[3]       = { wbf1, wbf2, wbf3 };
    const float*          bs[3]       = { b1, b2, b3 };

    for (int L = 0; L < 3; L++) {
        gemm128_wmma<<<gemm_blocks, 256, lds128, stream>>>(
            layer_in[L], Ws[L], nullptr, buf1, N_NODES, 128, 0);
        hipMemsetAsync(buf2, 0, (size_t)N_NODES * HID * 4, stream);
        edge_agg_kernel<<<agg_blocks, 256, 0, stream>>>(src, dst, dinv, buf1, buf2, N_EDGES);
        selfloop_bias_kernel<<<elem_blocks, 256, 0, stream>>>(
            buf1, dinv, bs[L], buf2, N_NODES, (L < 2) ? 1 : 0);
    }

    // global mean + max pooling
    pool_init_kernel<<<(N_GRAPHS * HID + 255) / 256, 256, 0, stream>>>(sums, cnt, mx);
    pool_kernel<<<N_NODES, 128, 0, stream>>>(buf2, batc, sums, cnt, mx, N_NODES);
    pool_finalize_kernel<<<(N_GRAPHS * 256 + 255) / 256, 256, 0, stream>>>(sums, cnt, mx, g);

    // classifier: relu(g @ Wc1 + bc1) then @ Wc2 + bc2
    const size_t lds256 = (size_t)(256 * 128 + 128 * 256) * 2;       // 128 KB
    gemm128_wmma<<<(N_GRAPHS + 127) / 128, 256, lds256, stream>>>(
        g, wbfc1, bc1, c1, N_GRAPHS, 256, 1);
    final_linear_kernel<<<(N_GRAPHS * 10 + 255) / 256, 256, 0, stream>>>(
        c1, Wc2, bc2, (float*)d_out);
}
